// _Interactions_66254165508965
// MI455X (gfx1250) — matrix operand order repaired and checked by reading, not compiled, not run on hardware
//
#include <hip/hip_runtime.h>

#define N_NODES 100000
#define N_EDGES 1600000
#define FILTERS 64
#define ZROWS 133          // 2*FILTERS + 5

typedef __attribute__((ext_vector_type(16))) __bf16 v16bf;
typedef __attribute__((ext_vector_type(8)))  float  v8f;
typedef unsigned int u32x4 __attribute__((ext_vector_type(4)));

union V16 { v16bf v; unsigned short u[16]; u32x4 q[2]; };
union V8F { v8f v; float f[8]; };

// fp32 -> bf16 round-to-nearest-even
__device__ __forceinline__ unsigned short f2bf(float f) {
  unsigned int u = __float_as_uint(f);
  return (unsigned short)((u + 0x7FFFu + ((u >> 16) & 1u)) >> 16);
}
__device__ __forceinline__ unsigned int pack2(float a, float b) {
  return (unsigned int)f2bf(a) | ((unsigned int)f2bf(b) << 16);
}

// 32-bit LDS-relative address of a generic pointer into __shared__
__device__ __forceinline__ unsigned lds_addr32(const void* p) {
  return (unsigned)(size_t)(const __attribute__((address_space(3))) void*)p;
}

// CDNA5 async copy: 16B global -> LDS, tracked by ASYNCcnt (bypasses VGPRs)
__device__ __forceinline__ void async_load_b128(unsigned lds_off, const void* gaddr) {
  asm volatile("global_load_async_to_lds_b128 %0, %1, off"
               :: "v"(lds_off), "v"((unsigned long long)(size_t)gaddr)
               : "memory");
}
__device__ __forceinline__ void wait_async0() {
  asm volatile("s_wait_asynccnt 0x0" ::: "memory");
}

// Build a B-matrix (32x16, KxN) bf16 fragment per ISA layout:
// lane%16 = column N; element j -> K = kb*32 + (lane/16)*8 + j (j<8), +8 more for j>=8
__device__ __forceinline__ v16bf load_bfrag(const float* __restrict__ w,
                                            int kb, int khalf, int ncol) {
  V16 r;
  const int kbase = kb * 32 + khalf * 8;
#pragma unroll
  for (int j = 0; j < 16; ++j)
    r.u[j] = f2bf(w[(kbase + j + (j >= 8 ? 8 : 0)) * FILTERS + ncol]);
  return r.v;
}

// ---------------- lin0: out = relu(h @ W + b), tiles of 16 nodes ----------------
__global__ __launch_bounds__(128)
void lin0_kernel(const float* __restrict__ h, const float* __restrict__ w,
                 const float* __restrict__ b, float* __restrict__ out) {
  const int t = threadIdx.x, lane = t & 31, wv = t >> 5;
  const int nloc = lane & 15, khalf = lane >> 4, nglob = wv * 16 + nloc;

  v16bf B0 = load_bfrag(w, 0, khalf, nglob);
  v16bf B1 = load_bfrag(w, 1, khalf, nglob);
  const float bias = b[nglob];

  const int ngroups = N_NODES / 16;
  for (int g = blockIdx.x; g < ngroups; g += gridDim.x) {
    const int row = g * 16 + nloc;   // A row = M = lane%16
    V8F c;
#pragma unroll
    for (int v = 0; v < 8; ++v) c.f[v] = bias;
#pragma unroll
    for (int kb = 0; kb < 2; ++kb) {
      // A fragment: element j -> K = kb*32 + khalf*8 + j ; element 8+j -> +16
      const float* hp = h + (size_t)row * 64 + kb * 32 + khalf * 8;
      V16 a;
#pragma unroll
      for (int j = 0; j < 8; ++j) { a.u[j] = f2bf(hp[j]); a.u[8 + j] = f2bf(hp[16 + j]); }
      c.v = __builtin_amdgcn_wmma_f32_16x16x32_bf16(false, a.v, false, (kb ? B1 : B0),
                                                    (short)0, c.v, false, false);
    }
    // one 64-bit base per group; 8 stores with immediate offsets
    float* op = out + (size_t)(g * 16 + 8 * khalf) * 64 + nglob;
#pragma unroll
    for (int v = 0; v < 8; ++v) op[v * 64] = fmaxf(c.f[v], 0.f);
  }
}

// ---------------- shortcut: e = relu(edge_attr @ Ws + bs), [E,5] ----------------
__global__ void shortcut_kernel(const float* __restrict__ attr, const float* __restrict__ w,
                                const float* __restrict__ b, float* __restrict__ e) {
  const int i = blockIdx.x * blockDim.x + threadIdx.x;
  if (i >= N_EDGES) return;
  float a[5];
#pragma unroll
  for (int g = 0; g < 5; ++g) a[g] = attr[(size_t)i * 5 + g];
#pragma unroll
  for (int n = 0; n < 5; ++n) {
    float s = b[n];
#pragma unroll
    for (int g = 0; g < 5; ++g) s = fmaf(a[g], w[g * 5 + n], s);
    e[(size_t)i * 5 + n] = fmaxf(s, 0.f);
  }
}

__global__ void zero_kernel(float4* __restrict__ p, int n4) {
  for (int i = blockIdx.x * blockDim.x + threadIdx.x; i < n4; i += gridDim.x * blockDim.x)
    p[i] = make_float4(0.f, 0.f, 0.f, 0.f);
}

// fp32 -> bf16 mirror of the node features (pairs -> packed u32 stores)
__global__ void cvt_bf16_kernel(const float* __restrict__ xin, unsigned int* __restrict__ xb, int n2) {
  for (int i = blockIdx.x * blockDim.x + threadIdx.x; i < n2; i += gridDim.x * blockDim.x) {
    float2 a = ((const float2*)xin)[i];
    xb[i] = pack2(a.x, a.y);
  }
}

// x = relu(x + agg); optionally also refresh the bf16 mirror
__global__ void node_update(const float* __restrict__ xin, const float* __restrict__ agg,
                            float* __restrict__ xout, unsigned int* __restrict__ xbout, int n2) {
  for (int i = blockIdx.x * blockDim.x + threadIdx.x; i < n2; i += gridDim.x * blockDim.x) {
    float2 a = ((const float2*)xin)[i];
    float2 b = ((const float2*)agg)[i];
    float r0 = fmaxf(a.x + b.x, 0.f);
    float r1 = fmaxf(a.y + b.y, 0.f);
    ((float2*)xout)[i] = make_float2(r0, r1);
    if (xbout) xbout[i] = pack2(r0, r1);
  }
}

// ---------------- CGConv edge kernel: async gather -> WMMA -> gate -> scatter ----------------
__global__ __launch_bounds__(128)
void cgconv_edge_kernel(const unsigned short* __restrict__ xb,   // [N][64] bf16
                        const long long* __restrict__ ei,        // [2][E] int64
                        const float* __restrict__ e5,            // [E][5]
                        const float* __restrict__ wf, const float* __restrict__ bfv,
                        const float* __restrict__ ws, const float* __restrict__ bsv,
                        float* __restrict__ agg) {
  // bf16 node-row tiles, row stride 72 halfwords (36 uints, 144 B) -> conflict-free b128 loads
  __shared__ __align__(16) unsigned int tileU[2][16 * 36];  // [0]=x_j(src), [1]=x_i(dst)
  __shared__ float etile[16][5];
  __shared__ int   dsttile[16];

  const int t = threadIdx.x;
  const int lane = t & 31, wv = t >> 5;
  const int nloc = lane & 15, khalf = lane >> 4;
  const int nglob = wv * 16 + nloc;           // this wave owns output cols [16wv,16wv+16)

  // Loop-invariant register-resident B fragments (zf = xi@wf[:64] + xj@wf[64:128] + e@wf[128:])
  v16bf Bfi[2], Bfj[2], Bsi[2], Bsj[2];
#pragma unroll
  for (int kb = 0; kb < 2; ++kb) {
    Bfi[kb] = load_bfrag(wf,             kb, khalf, nglob);
    Bfj[kb] = load_bfrag(wf + 64 * 64,   kb, khalf, nglob);
    Bsi[kb] = load_bfrag(ws,             kb, khalf, nglob);
    Bsj[kb] = load_bfrag(ws + 64 * 64,   kb, khalf, nglob);
  }
  float wfe[5], wse[5];
#pragma unroll
  for (int g = 0; g < 5; ++g) {
    wfe[g] = wf[(128 + g) * 64 + nglob];
    wse[g] = ws[(128 + g) * 64 + nglob];
  }
  const float biasf = bfv[nglob], biass = bsv[nglob];

  // Per-thread gather role (loop-invariant): copies 32 B of one node row
  const int which = t >> 6;        // 0: x_j (src), 1: x_i (dst)
  const int r     = (t >> 2) & 15; // edge within group
  const int part  = t & 3;         // quarter of the 128-B bf16 row
  const unsigned ldsDst = lds_addr32(&tileU[which][r * 36 + part * 8]); // 16B-aligned

  const long long* srcp = ei;
  const long long* dstp = ei + N_EDGES;
  const int ngroups = N_EDGES / 16;

  for (int g = blockIdx.x; g < ngroups; g += gridDim.x) {
    // ---- async gather: 16 bf16 node rows x 2 tiles, no VGPR round-trip ----
    {
      const int eid  = g * 16 + r;
      const int node = (int)(which ? dstp[eid] : srcp[eid]);
      const char* gsrc = (const char*)xb + (size_t)node * 128 + part * 32;
      async_load_b128(ldsDst,      gsrc);
      async_load_b128(ldsDst + 16, gsrc + 16);
      if (t < 16) dsttile[t] = (int)dstp[g * 16 + t];
      if (t < 80) {
        const int rr = t / 5, gg = t - rr * 5;
        etile[rr][gg] = e5[(size_t)(g * 16 + rr) * 5 + gg];
      }
    }
    wait_async0();       // this wave's async copies landed in LDS
    __syncthreads();     // all waves' copies visible

    // ---- accumulator init: bias + e @ W_e (per C/D lane layout) ----
    V8F cf, cs;
#pragma unroll
    for (int v = 0; v < 8; ++v) {
      const int m = v + 8 * khalf;
      float sf = biasf, ss = biass;
#pragma unroll
      for (int gg = 0; gg < 5; ++gg) {
        const float ev = etile[m][gg];
        sf = fmaf(ev, wfe[gg], sf);
        ss = fmaf(ev, wse[gg], ss);
      }
      cf.f[v] = sf; cs.f[v] = ss;
    }

    // ---- z += x_j @ W_j + x_i @ W_i for both gates (8 WMMAs / group / wave) ----
    const unsigned int* rowj = &tileU[0][nloc * 36];
    const unsigned int* rowi = &tileU[1][nloc * 36];
#pragma unroll
    for (int kb = 0; kb < 2; ++kb) {
      const int c0 = kb * 16 + khalf * 4;  // uints: halves kb*32+khalf*8 and +16
      V16 aj, ai;
      aj.q[0] = *(const u32x4*)(rowj + c0);
      aj.q[1] = *(const u32x4*)(rowj + c0 + 8);
      ai.q[0] = *(const u32x4*)(rowi + c0);
      ai.q[1] = *(const u32x4*)(rowi + c0 + 8);
      cf.v = __builtin_amdgcn_wmma_f32_16x16x32_bf16(false, aj.v, false, Bfj[kb], (short)0, cf.v, false, false);
      cs.v = __builtin_amdgcn_wmma_f32_16x16x32_bf16(false, aj.v, false, Bsj[kb], (short)0, cs.v, false, false);
      cf.v = __builtin_amdgcn_wmma_f32_16x16x32_bf16(false, ai.v, false, Bfi[kb], (short)0, cf.v, false, false);
      cs.v = __builtin_amdgcn_wmma_f32_16x16x32_bf16(false, ai.v, false, Bsi[kb], (short)0, cs.v, false, false);
    }

    // ---- msg = sigmoid(zf)*softplus(zs); scatter-add into agg[dst] ----
#pragma unroll
    for (int v = 0; v < 8; ++v) {
      const float zf = cf.f[v], zs = cs.f[v];
      const float sg = 1.f / (1.f + __expf(-zf));
      const float sp = fmaxf(zs, 0.f) + __logf(1.f + __expf(-fabsf(zs)));
      const int node = dsttile[v + 8 * khalf];
      atomicAdd(agg + (size_t)node * 64 + nglob, sg * sp);
    }
    __syncthreads();   // LDS reuse fence before next group's gather
  }
}

extern "C" void kernel_launch(void* const* d_in, const int* in_sizes, int n_in,
                              void* d_out, int out_size, void* d_ws, size_t ws_size,
                              hipStream_t stream) {
  (void)in_sizes; (void)n_in; (void)out_size; (void)ws_size;
  const float*     h       = (const float*)d_in[0];
  const long long* ei      = (const long long*)d_in[1];
  const float*     eattr   = (const float*)d_in[3];
  const float*     lin0_w  = (const float*)d_in[5];
  const float*     lin0_b  = (const float*)d_in[6];
  const float*     short_w = (const float*)d_in[7];
  const float*     short_b = (const float*)d_in[8];
  const float*     cfw     = (const float*)d_in[9];
  const float*     cfb     = (const float*)d_in[10];
  const float*     csw     = (const float*)d_in[11];
  const float*     csb     = (const float*)d_in[12];

  char* ws = (char*)d_ws;
  float*          x   = (float*)ws;                                     // N*64 f32   (25.6 MB)
  unsigned short* xb  = (unsigned short*)(ws + (size_t)N_NODES*64*4);   // N*64 bf16  (12.8 MB)
  float*          e5  = (float*)(ws + (size_t)N_NODES*64*6);            // E*5 f32    (32 MB)
  float*          agg = (float*)(ws + (size_t)N_NODES*64*6 + (size_t)N_EDGES*5*4);
  float*          out = (float*)d_out;

  const int n4 = N_NODES * 64 / 4;
  const int n2 = N_NODES * 64 / 2;

  lin0_kernel<<<1536, 128, 0, stream>>>(h, lin0_w, lin0_b, x);
  cvt_bf16_kernel<<<2048, 256, 0, stream>>>(x, (unsigned int*)xb, n2);
  shortcut_kernel<<<(N_EDGES + 255) / 256, 256, 0, stream>>>(eattr, short_w, short_b, e5);

  // layer 0
  zero_kernel<<<2048, 256, 0, stream>>>((float4*)agg, n4);
  cgconv_edge_kernel<<<2048, 128, 0, stream>>>(xb, ei, e5, cfw, cfb, csw, csb, agg);
  node_update<<<2048, 256, 0, stream>>>(x, agg, x, (unsigned int*)xb, n2);

  // layer 1 -> d_out
  zero_kernel<<<2048, 256, 0, stream>>>((float4*)agg, n4);
  cgconv_edge_kernel<<<2048, 128, 0, stream>>>(xb, ei, e5, cfw + ZROWS * 64, cfb + 64,
                                               csw + ZROWS * 64, csb + 64, agg);
  node_update<<<2048, 256, 0, stream>>>(x, agg, out, (unsigned int*)nullptr, n2);
}
